// CompositionalLayer_987842478192
// MI455X (gfx1250) — matrix operand order repaired and checked by reading, not compiled, not run on hardware
//
#include <hip/hip_runtime.h>
#include <hip/hip_bf16.h>
#include <math.h>

// ---------------------------------------------------------------------------
// cosine-similarity softmax vs. vocabulary, CDNA5 (gfx1250) WMMA bf16-split
//   x:     (32768, 576) fp32      vocab: (2048, 576) fp32
//   out:   softmax_v( (x.v) / (|x||v| + 1e-8) )   (32768, 2048) fp32
// GEMM runs as hi/lo bf16 split (3x v_wmma_f32_16x16x32_bf16 per K=32 chunk),
// fp32 accumulate -> ~22 effective mantissa bits.
// M-tile = 32 rows/block: each B fragment pair feeds 6 WMMAs (2 M-subtiles),
// halving vocab L2 traffic vs a 16-row tile. Softmax stays fused in-block.
// ---------------------------------------------------------------------------

typedef __attribute__((ext_vector_type(16))) __bf16 v16bf;
typedef __attribute__((ext_vector_type(8)))  float  v8f;

#define D_DIM     576          // 64*3*3
#define VCAB      2048
#define ROWTILE   32           // x rows per block (2 M-subtiles of 16)
#define BLOCKT    512          // 16 waves of 32
#define NWAVES    16
#define COLS_PER_WAVE (VCAB / NWAVES)     // 128
#define NTILES    (COLS_PER_WAVE / 16)    // 8 column tiles per wave
#define KCHUNKS   (D_DIM / 32)            // 18
#define LDA       (D_DIM + 8)             // padded LDS row stride (bank-safe)

union FragBF {
  v16bf v;
  uint4 q[2];
  unsigned short s[16];
};

__device__ __forceinline__ unsigned short f32_to_bf16_rne(float f) {
  unsigned int u = __float_as_uint(f);
  u += 0x7FFFu + ((u >> 16) & 1u);         // round-to-nearest-even
  return (unsigned short)(u >> 16);
}
__device__ __forceinline__ float bf16_bits_to_f32(unsigned short h) {
  return __uint_as_float(((unsigned int)h) << 16);
}

// ---- kernel 1: vocab -> bf16 hi/lo + vocab-row norms ----------------------
__global__ void __launch_bounds__(256)
vocab_prep_kernel(const float* __restrict__ vocab,
                  unsigned short* __restrict__ vhi,
                  unsigned short* __restrict__ vlo,
                  float* __restrict__ vnorm) {
  const int v    = blockIdx.x * 8 + (threadIdx.x >> 5);   // one wave / row
  const int lane = threadIdx.x & 31;
  const float* row = vocab + (size_t)v * D_DIM;
  float ss = 0.f;
  for (int k = lane; k < D_DIM; k += 32) {
    float val = row[k];
    unsigned short hb = f32_to_bf16_rne(val);
    vhi[(size_t)v * D_DIM + k] = hb;
    vlo[(size_t)v * D_DIM + k] = f32_to_bf16_rne(val - bf16_bits_to_f32(hb));
    ss += val * val;
  }
  #pragma unroll
  for (int m = 16; m >= 1; m >>= 1) ss += __shfl_xor(ss, m, 32);
  if (lane == 0) vnorm[v] = sqrtf(ss);
}

// ---- kernel 2: fused GEMM + cosine + row softmax --------------------------
__global__ void __launch_bounds__(BLOCKT)
cos_softmax_kernel(const float* __restrict__ x,
                   const unsigned short* __restrict__ vhi,
                   const unsigned short* __restrict__ vlo,
                   const float* __restrict__ vnorm,
                   float* __restrict__ out) {
  __shared__ unsigned short Ahi[ROWTILE][LDA];   // 36.5 KB
  __shared__ unsigned short Alo[ROWTILE][LDA];   // 36.5 KB
  __shared__ float xnorm[ROWTILE];
  __shared__ float red[NWAVES * ROWTILE];        // 2 KB
  __shared__ float fin[ROWTILE];

  const int tid  = threadIdx.x;
  const int lane = tid & 31;
  const int wave = tid >> 5;
  const int half = lane >> 4;      // 0: lanes 0-15, 1: lanes 16-31
  const int l16  = lane & 15;
  const int row0 = blockIdx.x * ROWTILE;

  // ---- stage x tile to LDS as bf16 hi/lo; row norms (16 threads / row) ----
  {
    const int arow = tid >> 4;     // 0..31
    const int kl   = tid & 15;
    const float* xr = x + (size_t)(row0 + arow) * D_DIM;
    float ss = 0.f;
    for (int k = kl; k < D_DIM; k += 16) {
      float v = xr[k];
      unsigned short hb = f32_to_bf16_rne(v);
      Ahi[arow][k] = hb;
      Alo[arow][k] = f32_to_bf16_rne(v - bf16_bits_to_f32(hb));
      ss += v * v;
    }
    #pragma unroll
    for (int m = 8; m >= 1; m >>= 1) ss += __shfl_xor(ss, m, 32);
    if (kl == 0) xnorm[arow] = sqrtf(ss);
  }
  __syncthreads();

  // ---- GEMM: wave owns 32 rows x 128 cols, K = 576 ----
  const int colbase = wave * COLS_PER_WAVE;
  v8f acc[2][NTILES];              // [M-subtile][col tile] : 128 VGPRs
  #pragma unroll
  for (int s = 0; s < 2; ++s)
    #pragma unroll
    for (int t = 0; t < NTILES; ++t)
      #pragma unroll
      for (int r = 0; r < 8; ++r) acc[s][t][r] = 0.f;

  // per-wave B base: col = colbase + l16 (+t*16), K offset = kc*32 + half*8
  const size_t bbase = (size_t)(colbase + l16) * D_DIM + half * 8;
  const unsigned short* __restrict__ pbh = vhi + bbase;
  const unsigned short* __restrict__ pbl = vlo + bbase;

  for (int kc = 0; kc < KCHUNKS; ++kc) {
    const int k0 = kc * 32;
    // A fragments (ISA 16-bit A 16x32 layout: lane<16 holds K{0..7,16..23})
    FragBF ah0, al0, ah1, al1;
    {
      const int kb = k0 + half * 8;
      ah0.q[0] = *(const uint4*)&Ahi[l16][kb];
      ah0.q[1] = *(const uint4*)&Ahi[l16][kb + 16];
      al0.q[0] = *(const uint4*)&Alo[l16][kb];
      al0.q[1] = *(const uint4*)&Alo[l16][kb + 16];
      ah1.q[0] = *(const uint4*)&Ahi[16 + l16][kb];
      ah1.q[1] = *(const uint4*)&Ahi[16 + l16][kb + 16];
      al1.q[0] = *(const uint4*)&Alo[16 + l16][kb];
      al1.q[1] = *(const uint4*)&Alo[16 + l16][kb + 16];
    }
    #pragma unroll
    for (int t = 0; t < NTILES; ++t) {
      // B fragment (same layout as A: lanes 0-15 K{0..7,16..23}, lanes 16-31 +8);
      // constant per-tile byte offset folds into the load immediate.
      const int eoff = k0 + t * 16 * D_DIM;
      FragBF bh, bl;
      bh.q[0] = ((const uint4*)(pbh + eoff))[0];
      bh.q[1] = ((const uint4*)(pbh + eoff))[2];
      bl.q[0] = ((const uint4*)(pbl + eoff))[0];
      bl.q[1] = ((const uint4*)(pbl + eoff))[2];
      // M-subtile 0 (rows 0-15)
      acc[0][t] = __builtin_amdgcn_wmma_f32_16x16x32_bf16(
          false, ah0.v, false, bh.v, (short)0, acc[0][t], false, false);
      acc[0][t] = __builtin_amdgcn_wmma_f32_16x16x32_bf16(
          false, ah0.v, false, bl.v, (short)0, acc[0][t], false, false);
      acc[0][t] = __builtin_amdgcn_wmma_f32_16x16x32_bf16(
          false, al0.v, false, bh.v, (short)0, acc[0][t], false, false);
      // M-subtile 1 (rows 16-31) reuses the same B fragments
      acc[1][t] = __builtin_amdgcn_wmma_f32_16x16x32_bf16(
          false, ah1.v, false, bh.v, (short)0, acc[1][t], false, false);
      acc[1][t] = __builtin_amdgcn_wmma_f32_16x16x32_bf16(
          false, ah1.v, false, bl.v, (short)0, acc[1][t], false, false);
      acc[1][t] = __builtin_amdgcn_wmma_f32_16x16x32_bf16(
          false, al1.v, false, bh.v, (short)0, acc[1][t], false, false);
    }
  }

  // ---- cosine scale + row max (C layout: VGPR r -> row r + half*8) ----
  float xr[2][8];
  #pragma unroll
  for (int s = 0; s < 2; ++s)
    #pragma unroll
    for (int r = 0; r < 8; ++r) xr[s][r] = xnorm[s * 16 + half * 8 + r];

  float rmax[2][8];
  #pragma unroll
  for (int s = 0; s < 2; ++s)
    #pragma unroll
    for (int r = 0; r < 8; ++r) rmax[s][r] = -1e30f;

  #pragma unroll
  for (int t = 0; t < NTILES; ++t) {
    float vn = vnorm[colbase + t * 16 + l16];
    #pragma unroll
    for (int s = 0; s < 2; ++s)
      #pragma unroll
      for (int r = 0; r < 8; ++r) {
        float c = acc[s][t][r] *
                  __builtin_amdgcn_rcpf(xr[s][r] * vn + 1e-8f);
        acc[s][t][r] = c;
        rmax[s][r] = fmaxf(rmax[s][r], c);
      }
  }
  #pragma unroll
  for (int m = 8; m >= 1; m >>= 1)
    #pragma unroll
    for (int s = 0; s < 2; ++s)
      #pragma unroll
      for (int r = 0; r < 8; ++r)
        rmax[s][r] = fmaxf(rmax[s][r], __shfl_xor(rmax[s][r], m, 32));
  if (l16 == 0) {
    #pragma unroll
    for (int s = 0; s < 2; ++s)
      #pragma unroll
      for (int r = 0; r < 8; ++r)
        red[wave * ROWTILE + s * 16 + half * 8 + r] = rmax[s][r];
  }
  __syncthreads();
  if (tid < ROWTILE) {
    float m = -1e30f;
    for (int w = 0; w < NWAVES; ++w) m = fmaxf(m, red[w * ROWTILE + tid]);
    fin[tid] = m;
  }
  __syncthreads();
  float fm[2][8];
  #pragma unroll
  for (int s = 0; s < 2; ++s)
    #pragma unroll
    for (int r = 0; r < 8; ++r) fm[s][r] = fin[s * 16 + half * 8 + r];

  // ---- exp + row sum (exp values kept in acc) ----
  float rsum[2][8];
  #pragma unroll
  for (int s = 0; s < 2; ++s)
    #pragma unroll
    for (int r = 0; r < 8; ++r) rsum[s][r] = 0.f;
  #pragma unroll
  for (int t = 0; t < NTILES; ++t)
    #pragma unroll
    for (int s = 0; s < 2; ++s)
      #pragma unroll
      for (int r = 0; r < 8; ++r) {
        float e = __expf(acc[s][t][r] - fm[s][r]);
        acc[s][t][r] = e;
        rsum[s][r] += e;
      }
  #pragma unroll
  for (int m = 8; m >= 1; m >>= 1)
    #pragma unroll
    for (int s = 0; s < 2; ++s)
      #pragma unroll
      for (int r = 0; r < 8; ++r) rsum[s][r] += __shfl_xor(rsum[s][r], m, 32);
  if (l16 == 0) {
    #pragma unroll
    for (int s = 0; s < 2; ++s)
      #pragma unroll
      for (int r = 0; r < 8; ++r)
        red[wave * ROWTILE + s * 16 + half * 8 + r] = rsum[s][r];
  }
  __syncthreads();
  if (tid < ROWTILE) {
    float ssum = 0.f;
    for (int w = 0; w < NWAVES; ++w) ssum += red[w * ROWTILE + tid];
    fin[tid] = 1.0f / ssum;
  }
  __syncthreads();
  float rs[2][8];
  #pragma unroll
  for (int s = 0; s < 2; ++s)
    #pragma unroll
    for (int r = 0; r < 8; ++r) rs[s][r] = fin[s * 16 + half * 8 + r];

  // ---- store (16 consecutive lanes -> 16 consecutive cols) ----
  #pragma unroll
  for (int t = 0; t < NTILES; ++t) {
    const int col = colbase + t * 16 + l16;
    #pragma unroll
    for (int s = 0; s < 2; ++s)
      #pragma unroll
      for (int r = 0; r < 8; ++r) {
        out[(size_t)(row0 + s * 16 + half * 8 + r) * VCAB + col] =
            acc[s][t][r] * rs[s][r];
      }
  }
}

// ---------------------------------------------------------------------------
extern "C" void kernel_launch(void* const* d_in, const int* in_sizes, int n_in,
                              void* d_out, int out_size, void* d_ws, size_t ws_size,
                              hipStream_t stream) {
  const float* x     = (const float*)d_in[0];   // (8*4096, 576) fp32
  const float* vocab = (const float*)d_in[1];   // (2048, 576)   fp32
  float* out = (float*)d_out;                   // (8*4096, 2048) fp32

  // workspace layout: vocab hi (2.25MB) | vocab lo (2.25MB) | vnorm (8KB)
  unsigned short* vhi = (unsigned short*)d_ws;
  unsigned short* vlo = vhi + (size_t)VCAB * D_DIM;
  float* vnorm = (float*)(vlo + (size_t)VCAB * D_DIM);

  const int rows = in_sizes[0] / D_DIM;         // 32768

  vocab_prep_kernel<<<VCAB / 8, 256, 0, stream>>>(vocab, vhi, vlo, vnorm);
  cos_softmax_kernel<<<rows / ROWTILE, BLOCKT, 0, stream>>>(x, vhi, vlo, vnorm, out);
}
